// ACTReasoningGNN_83013127897699
// MI455X (gfx1250) — matrix operand order repaired and checked by reading, not compiled, not run on hardware
//
#include <hip/hip_runtime.h>
#include <math.h>

// ---------------- problem constants (from reference) ----------------
#define N_NODES 10000
#define D_IN    384
#define HID     256
#define NHEAD   4
#define OUT_CH  128
#define E_EDGES 160000
#define HD      (NHEAD * HID)      // 1024
#define CATW    (D_IN + HID)       // 640
#define ET      (E_EDGES + N_NODES)

typedef __bf16 bf16_t;
typedef __attribute__((ext_vector_type(16))) __bf16 v16bf;
typedef __attribute__((ext_vector_type(8)))  __bf16 v8bf;
typedef __attribute__((ext_vector_type(8)))  float  v8f;

// ---------------- helpers ----------------
__device__ __forceinline__ float lrelu02(float x) { return x > 0.f ? x : 0.2f * x; }

__device__ __forceinline__ void atomicMaxF(float* a, float v) {
    if (v >= 0.f) atomicMax((int*)a, __float_as_int(v));
    else          atomicMin((unsigned int*)a, __float_as_uint(v));
}

// A operand (16x32 bf16, MxK): lane L<16 -> M=L, K in {hi*8..hi*8+7} U {16+hi*8..16+hi*8+7}
__device__ __forceinline__ v16bf load_a(const bf16_t* __restrict__ p, int k0, int hi) {
    v8bf lo = *(const v8bf*)(p + k0 + hi * 8);
    v8bf hh = *(const v8bf*)(p + k0 + 16 + hi * 8);
    v16bf r;
#pragma unroll
    for (int i = 0; i < 8; i++) { r[i] = lo[i]; r[i + 8] = hh[i]; }
    return r;
}

// B operand (32x16 bf16, KxN), loaded from pre-transposed Bt[N][K]:
// lane holds column n = lane&15, K in {hi*8..hi*8+7} U {16+hi*8..16+hi*8+7}
__device__ __forceinline__ v16bf load_b(const bf16_t* __restrict__ p, int k0, int hi) {
    v8bf lo = *(const v8bf*)(p + k0 + hi * 8);
    v8bf hh = *(const v8bf*)(p + k0 + 16 + hi * 8);
    v16bf r;
#pragma unroll
    for (int i = 0; i < 8; i++) { r[i] = lo[i]; r[i + 8] = hh[i]; }
    return r;
}

// ---------------- kernels ----------------

// Fill a float buffer with a constant (amax=-inf, denom=0, accum=0).
__global__ void k_fill(float* __restrict__ p, int n, float v) {
    int t = blockIdx.x * blockDim.x + threadIdx.x;
    if (t < n) p[t] = v;
}

// Build bf16 concat [N, 640] = [x | emb_node[node_type]]
__global__ void k_concat(const float* __restrict__ x, const int* __restrict__ ntype,
                         const float* __restrict__ emb, bf16_t* __restrict__ cat) {
    int row = blockIdx.x;
    const float* xr = x + (size_t)row * D_IN;
    const float* er = emb + (size_t)ntype[row] * HID;
    bf16_t* o = cat + (size_t)row * CATW;
    for (int c = threadIdx.x; c < CATW; c += blockDim.x) {
        float v = (c < D_IN) ? xr[c] : er[c - D_IN];
        o[c] = (bf16_t)v;
    }
}

// Convert + transpose fp32 W[K][Nc] -> bf16 Wt[Nc][K]  (weights, small, once per layer)
__global__ void k_cvtT(const float* __restrict__ W, bf16_t* __restrict__ Wt, int K, int Nc) {
    int o = blockIdx.x * blockDim.x + threadIdx.x;
    if (o >= K * Nc) return;
    int n = o / K;
    int k = o - n * K;
    Wt[o] = (bf16_t)W[(size_t)k * Nc + n];
}

// WMMA GEMM: C[M=10000, Nc] = A[M,K]bf16 * Bt[Nc,K]bf16^T (+bias), f32 accumulate.
// Wave computes a 16x32 strip (2 accumulators); 8 waves/block share the M tile.
// grid = ((Nc+255)/256, 625), block = 256.
__global__ void k_gemm_bf16(const bf16_t* __restrict__ A, const bf16_t* __restrict__ Bt,
                            const float* __restrict__ bias, float* __restrict__ Cf,
                            bf16_t* __restrict__ Cb, int Nc, int K) {
    const int lane    = threadIdx.x & 31;
    const int wave    = threadIdx.x >> 5;
    const int colBase = (blockIdx.x * 8 + wave) * 32;
    if (colBase >= Nc) return;                      // wave-uniform: EXEC all-ones for WMMA
    const int hi      = lane >> 4;
    const int l15     = lane & 15;
    const int rowBase = blockIdx.y * 16;

    const bf16_t* arow  = A  + (size_t)(rowBase + l15) * K;
    const bf16_t* brow0 = Bt + (size_t)(colBase + l15) * K;
    const bf16_t* brow1 = Bt + (size_t)(colBase + 16 + l15) * K;

    v8f acc0 = {};
    v8f acc1 = {};
    for (int k0 = 0; k0 < K; k0 += 32) {
        v16bf av  = load_a(arow,  k0, hi);
        v16bf bv0 = load_b(brow0, k0, hi);
        v16bf bv1 = load_b(brow1, k0, hi);
        acc0 = __builtin_amdgcn_wmma_f32_16x16x32_bf16(false, av, false, bv0,
                                                       (short)0, acc0, false, false);
        acc1 = __builtin_amdgcn_wmma_f32_16x16x32_bf16(false, av, false, bv1,
                                                       (short)0, acc1, false, false);
    }

    float b0 = bias ? bias[colBase + l15]      : 0.f;
    float b1 = bias ? bias[colBase + 16 + l15] : 0.f;
#pragma unroll
    for (int r = 0; r < 8; r++) {
        int row   = rowBase + r + 8 * hi;            // C/D layout: VGPR r -> M = r + 8*hi
        size_t o0 = (size_t)row * Nc + colBase + l15;
        float v0 = acc0[r] + b0;
        float v1 = acc1[r] + b1;
        if (Cf) { Cf[o0] = v0; Cf[o0 + 16] = v1; }
        if (Cb) { Cb[o0] = (bf16_t)v0; Cb[o0 + 16] = (bf16_t)v1; }
    }
}

// a_s[n,h] = dot(xh[n,h,:], att_s[h,:]); same for a_d. grid=N, block=128 (wave==head).
__global__ void k_attn(const float* __restrict__ xh, const float* __restrict__ att_s,
                       const float* __restrict__ att_d, float* __restrict__ a_s,
                       float* __restrict__ a_d) {
    int n    = blockIdx.x;
    int head = threadIdx.x >> 5;
    int lane = threadIdx.x & 31;
    const float* xr = xh + (size_t)n * HD + head * HID;
    const float* as = att_s + head * HID;
    const float* ad = att_d + head * HID;
    float ss = 0.f, sd = 0.f;
#pragma unroll
    for (int i = 0; i < 8; i++) {
        int c = lane * 8 + i;
        float v = xr[c];
        ss += v * as[c];
        sd += v * ad[c];
    }
    for (int off = 16; off > 0; off >>= 1) {
        ss += __shfl_down(ss, off, 32);
        sd += __shfl_down(sd, off, 32);
    }
    if (lane == 0) {
        a_s[n * NHEAD + head] = ss;
        a_d[n * NHEAD + head] = sd;
    }
}

// Pass 1: alpha_raw = leaky_relu(a_s[src]+a_d[dst]); atomic segment-max over dst.
__global__ void k_edge_max(const int* __restrict__ src, const int* __restrict__ dst,
                           const float* __restrict__ a_s, const float* __restrict__ a_d,
                           float* __restrict__ alpha, float* __restrict__ amax) {
    int e = blockIdx.x * blockDim.x + threadIdx.x;
    if (e >= ET) return;
    int s = (e < E_EDGES) ? src[e] : (e - E_EDGES);
    int d = (e < E_EDGES) ? dst[e] : (e - E_EDGES);
    float4 vs = *(const float4*)(a_s + 4 * (size_t)s);
    float4 vd = *(const float4*)(a_d + 4 * (size_t)d);
    float4 al;
    al.x = lrelu02(vs.x + vd.x);
    al.y = lrelu02(vs.y + vd.y);
    al.z = lrelu02(vs.z + vd.z);
    al.w = lrelu02(vs.w + vd.w);
    *(float4*)(alpha + 4 * (size_t)e) = al;
    float* am = amax + 4 * (size_t)d;
    atomicMaxF(am + 0, al.x);
    atomicMaxF(am + 1, al.y);
    atomicMaxF(am + 2, al.z);
    atomicMaxF(am + 3, al.w);
}

// Pass 2: alpha = exp(alpha - amax[dst]); atomic segment-sum -> denom.
__global__ void k_edge_expsum(const int* __restrict__ dst, float* __restrict__ alpha,
                              const float* __restrict__ amax, float* __restrict__ denom) {
    int e = blockIdx.x * blockDim.x + threadIdx.x;
    if (e >= ET) return;
    int d = (e < E_EDGES) ? dst[e] : (e - E_EDGES);
    float4 al = *(const float4*)(alpha + 4 * (size_t)e);
    float4 mx = *(const float4*)(amax + 4 * (size_t)d);
    al.x = expf(al.x - mx.x);
    al.y = expf(al.y - mx.y);
    al.z = expf(al.z - mx.z);
    al.w = expf(al.w - mx.w);
    *(float4*)(alpha + 4 * (size_t)e) = al;
    float* dn = denom + 4 * (size_t)d;
    atomicAdd(dn + 0, al.x);
    atomicAdd(dn + 1, al.y);
    atomicAdd(dn + 2, al.z);
    atomicAdd(dn + 3, al.w);
}

// Scatter: accum[dst] += xh[src] * (alpha / (denom[dst]+1e-16)). One block per edge,
// 256 threads x float4 = 1024 ch; gather/atomics are L2-resident (xh = 40MB < 192MB L2).
__global__ void k_scatter(const int* __restrict__ src, const int* __restrict__ dst,
                          const float* __restrict__ xh, const float* __restrict__ alpha,
                          const float* __restrict__ denom, float* __restrict__ accum) {
    int e = blockIdx.x;
    int s = (e < E_EDGES) ? src[e] : (e - E_EDGES);
    int d = (e < E_EDGES) ? dst[e] : (e - E_EDGES);
    int j  = threadIdx.x;
    int c0 = j * 4;
    int h  = j >> 6;                         // c0 / 256
    float w = alpha[4 * (size_t)e + h] / (denom[4 * (size_t)d + h] + 1e-16f);
    float4 xv = *(const float4*)(xh + (size_t)s * HD + c0);
    float* o = accum + (size_t)d * HD + c0;
    atomicAdd(o + 0, xv.x * w);
    atomicAdd(o + 1, xv.y * w);
    atomicAdd(o + 2, xv.z * w);
    atomicAdd(o + 3, xv.w * w);
}

// h = elu?(accum + bias) -> bf16 (feeds next GEMM's A operand)
__global__ void k_bias_elu_cvt(const float* __restrict__ accum, const float* __restrict__ bias,
                               bf16_t* __restrict__ hout, int elu) {
    int t = blockIdx.x * blockDim.x + threadIdx.x;
    if (t >= N_NODES * HD) return;
    float v = accum[t] + bias[t & (HD - 1)];
    if (elu) v = (v > 0.f) ? v : (expf(v) - 1.f);
    hout[t] = (bf16_t)v;
}

// ---------------- driver ----------------
extern "C" void kernel_launch(void* const* d_in, const int* in_sizes, int n_in,
                              void* d_out, int out_size, void* d_ws, size_t ws_size,
                              hipStream_t stream) {
    const float* x      = (const float*)d_in[0];
    const int*   eidx   = (const int*)d_in[1];
    const int*   src    = eidx;
    const int*   dst    = eidx + E_EDGES;
    const int*   ntype  = (const int*)d_in[2];
    const float* emb    = (const float*)d_in[3];
    const float* w_in   = (const float*)d_in[4];
    const float* b_in   = (const float*)d_in[5];
    const float* w_out  = (const float*)d_in[6];
    const float* b_out  = (const float*)d_in[7];
    const float* lin[3]   = {(const float*)d_in[8],  (const float*)d_in[12], (const float*)d_in[16]};
    const float* att_s[3] = {(const float*)d_in[9],  (const float*)d_in[13], (const float*)d_in[17]};
    const float* att_d[3] = {(const float*)d_in[10], (const float*)d_in[14], (const float*)d_in[18]};
    const float* lbias[3] = {(const float*)d_in[11], (const float*)d_in[15], (const float*)d_in[19]};

    // workspace carve-up (256B aligned), ~108 MB total
    char* p = (char*)d_ws;
    auto carve = [&](size_t bytes) {
        void* r = (void*)p;
        p += (bytes + 255) & ~(size_t)255;
        return r;
    };
    bf16_t* h_bf  = (bf16_t*)carve((size_t)N_NODES * HD * 2);   // activations, bf16
    float*  xh    = (float*) carve((size_t)N_NODES * HD * 4);   // projected features, f32
    float*  accum = (float*) carve((size_t)N_NODES * HD * 4);   // segment-sum accumulator
    bf16_t* Wt    = (bf16_t*)carve((size_t)HD * HD * 2);        // transposed bf16 weight
    float*  a_s   = (float*) carve((size_t)N_NODES * NHEAD * 4);
    float*  a_d   = (float*) carve((size_t)N_NODES * NHEAD * 4);
    float*  amax  = (float*) carve((size_t)N_NODES * NHEAD * 4);
    float*  denom = (float*) carve((size_t)N_NODES * NHEAD * 4);
    float*  alpha = (float*) carve((size_t)ET * NHEAD * 4);

    const int MT = N_NODES / 16;   // 625 M-tiles, exact
    const int N4 = N_NODES * NHEAD;
    const int NHDsz = N_NODES * HD;

    // Stage 0: concat(x, emb[node_type]) -> bf16, then h0 = cat @ w_in + b_in (bf16 out)
    bf16_t* cat = (bf16_t*)xh;  // alias: xh is free during input stage
    k_concat<<<N_NODES, 320, 0, stream>>>(x, ntype, emb, cat);
    k_cvtT<<<(CATW * HID + 255) / 256, 256, 0, stream>>>(w_in, Wt, CATW, HID);
    k_gemm_bf16<<<dim3((HID + 255) / 256, MT), 256, 0, stream>>>(
        cat, Wt, b_in, nullptr, h_bf, HID, CATW);

    const int in_ch[3] = {HID, HD, HD};
    for (int i = 0; i < 3; i++) {
        // xh = h @ lin_i   (no bias pre-aggregation, matches reference)
        k_cvtT<<<(in_ch[i] * HD + 255) / 256, 256, 0, stream>>>(lin[i], Wt, in_ch[i], HD);
        k_gemm_bf16<<<dim3((HD + 255) / 256, MT), 256, 0, stream>>>(
            h_bf, Wt, nullptr, xh, nullptr, HD, in_ch[i]);

        // attention scores
        k_attn<<<N_NODES, 128, 0, stream>>>(xh, att_s[i], att_d[i], a_s, a_d);

        // softmax over destination segments + message scatter
        k_fill<<<(N4 + 255) / 256, 256, 0, stream>>>(amax, N4, -INFINITY);
        k_fill<<<(N4 + 255) / 256, 256, 0, stream>>>(denom, N4, 0.f);
        k_fill<<<(NHDsz + 255) / 256, 256, 0, stream>>>(accum, NHDsz, 0.f);
        k_edge_max<<<(ET + 255) / 256, 256, 0, stream>>>(src, dst, a_s, a_d, alpha, amax);
        k_edge_expsum<<<(ET + 255) / 256, 256, 0, stream>>>(dst, alpha, amax, denom);
        k_scatter<<<ET, 256, 0, stream>>>(src, dst, xh, alpha, denom, accum);

        // bias (+ELU on layers 0,1) -> bf16 activations for next layer
        k_bias_elu_cvt<<<(NHDsz + 255) / 256, 256, 0, stream>>>(accum, lbias[i], h_bf, i < 2);
    }

    // out = h @ w_out + b_out  (f32 to d_out)
    k_cvtT<<<(HD * OUT_CH + 255) / 256, 256, 0, stream>>>(w_out, Wt, HD, OUT_CH);
    k_gemm_bf16<<<dim3((OUT_CH + 255) / 256, MT), 256, 0, stream>>>(
        h_bf, Wt, b_out, (float*)d_out, nullptr, OUT_CH, HD);
}
